// CRF_20298015441159
// MI455X (gfx1250) — matrix operand order repaired and checked by reading, not compiled, not run on hardware
//
#include <hip/hip_runtime.h>

// CRF log-likelihood, MI455X (gfx1250), wave32 + WMMA.
//
// Denominator scan step rewritten as a stabilized GEMM in probability space:
//   p = exp(s - rowmax);  q = p @ E  (E = exp(transitions), precomputed);
//   s' = rowmax + log(q) + emissions_step
// Hot loop: v_wmma_f32_16x16x32_f16 (f16 inputs, f32 accumulate).
//
// One wave owns a 16-batch tile: state s lives in registers in the WMMA C/D
// layout. The D->A layout transpose each step goes through a wave-private LDS
// buffer (f16, padded rows, 16B aligned -> ds_store_b16 / ds_load_b128); LDS
// is in-order per wave so an s_wait_dscnt fence is enough (no barriers).
//
// v3: use __builtin_amdgcn_exp2f / __builtin_amdgcn_logf (raw v_exp_f32 /
// v_log_f32) since this toolchain's HIP headers lack __exp2f/__log2f device
// intrinsics. Mask select stays an arithmetic FMA blend (no exec-mask
// branching around the log block).

typedef __attribute__((ext_vector_type(16))) _Float16 v16h;
typedef __attribute__((ext_vector_type(8)))  _Float16 h8;
typedef __attribute__((ext_vector_type(8)))  float    v8f;

#define CRF_B 512
#define CRF_S 1024
#define CRF_T 64
#define ROWB 144   // LDS row pitch: 64 f16 = 128B + 16B pad (bank spread, keeps 16B align)

#define LOG2E 1.4426950408889634f
#define LN2   0.6931471805599453f

__device__ __forceinline__ float fast_exp2(float x) { return __builtin_amdgcn_exp2f(x); }
__device__ __forceinline__ float fast_log2(float x) { return __builtin_amdgcn_logf(x); }

// ---------------------------------------------------------------------------
// Kernel 1: numerator (gold-path score), one thread per batch.
// ---------------------------------------------------------------------------
__global__ __launch_bounds__(256)
void crf_numerator(const float* __restrict__ em, const int* __restrict__ tags,
                   const int* __restrict__ mask, const float* __restrict__ start_t,
                   const float* __restrict__ end_t, const float* __restrict__ trans,
                   float* __restrict__ num) {
  int b = blockIdx.x * blockDim.x + threadIdx.x;
  if (b >= CRF_B) return;
  const size_t bST = (size_t)b * CRF_S * CRF_T;
  int tg0  = tags[(size_t)b * CRF_S];
  float sc = start_t[tg0] + em[bST + tg0];
  int prev = tg0, last = tg0;
  for (int s = 1; s < CRF_S; ++s) {
    int tg = tags[(size_t)b * CRF_S + s];
    int mk = mask[(size_t)b * CRF_S + s];
    float step = trans[prev * CRF_T + tg] + em[bST + (size_t)s * CRF_T + tg];
    sc += step * (float)mk;
    last = (mk > 0) ? tg : last;
    prev = tg;
  }
  num[b] = sc + end_t[last];
}

// ---------------------------------------------------------------------------
// Kernel 2: forward-algorithm scan via WMMA. 32 blocks x 32 threads = 1 wave
// per block, each wave owns batches [16*blockIdx, 16*blockIdx+16).
// ---------------------------------------------------------------------------
__global__ __launch_bounds__(32)
void crf_forward_scan(const float* __restrict__ em, const int* __restrict__ mask,
                      const float* __restrict__ start_t, const float* __restrict__ end_t,
                      const float* __restrict__ trans, float* __restrict__ den) {
  __shared__ __align__(16) unsigned char lds_raw[16 * ROWB];

  const int lane = threadIdx.x & 31;  // 0..31
  const int hw   = lane >> 4;         // half-wave: 0 or 1
  const int l16  = lane & 15;
  const int b0   = blockIdx.x * 16;   // batch tile base

  // --- Build B fragments of E = exp(transitions), resident for whole scan.
  // B fragments in the contiguous k order used here (K=32 x N=16): lane holds column n=l16;
  // lanes 0-15 hold K=0..15, lanes 16-31 hold K=16..31, element e <-> k offset e.
  v16h Bf[4][2];
  #pragma unroll
  for (int t = 0; t < 4; ++t) {
    #pragma unroll
    for (int h = 0; h < 2; ++h) {
      int n = t * 16 + l16;
      #pragma unroll
      for (int e = 0; e < 16; ++e) {
        int k = h * 32 + hw * 16 + e;
        Bf[t][h][e] = (_Float16)fast_exp2(trans[k * CRF_T + n] * LOG2E);
      }
    }
  }

  // --- State s in C/D layout: s[t][v] holds batch bl = v + 8*hw, j = t*16+l16.
  v8f s[4];
  #pragma unroll
  for (int t = 0; t < 4; ++t) {
    #pragma unroll
    for (int v = 0; v < 8; ++v) {
      int bl = v + 8 * hw;
      int j  = t * 16 + l16;
      s[t][v] = start_t[j] + em[((size_t)(b0 + bl)) * CRF_S * CRF_T + j];
    }
  }

  for (int step = 1; step < CRF_S; ++step) {
    // Emission tile for this step (D layout, 64B-coalesced per half-wave row).
    float ef[4][8];
    #pragma unroll
    for (int t = 0; t < 4; ++t)
      #pragma unroll
      for (int v = 0; v < 8; ++v) {
        int bl = v + 8 * hw;
        int j  = t * 16 + l16;
        ef[t][v] = em[((size_t)(b0 + bl)) * CRF_S * CRF_T + (size_t)step * CRF_T + j];
      }
    if (step + 1 < CRF_S)  // speculative prefetch of next step's segment
      __builtin_prefetch(&em[((size_t)(b0 + 8 * hw)) * CRF_S * CRF_T +
                             (size_t)(step + 1) * CRF_T + l16], 0, 0);

    // Mask as float blend factor (branchless; avoids exec-mask regions).
    float mf[8];
    #pragma unroll
    for (int v = 0; v < 8; ++v)
      mf[v] = (float)mask[(size_t)(b0 + v + 8 * hw) * CRF_S + step];

    // Per-batch max over j: xor-shuffles 1,2,4,8 stay inside each 16-lane half,
    // which is exactly the D-layout batch partition.
    float mx[8];
    #pragma unroll
    for (int v = 0; v < 8; ++v) {
      float m = fmaxf(fmaxf(s[0][v], s[1][v]), fmaxf(s[2][v], s[3][v]));
      m = fmaxf(m, __shfl_xor(m, 1));
      m = fmaxf(m, __shfl_xor(m, 2));
      m = fmaxf(m, __shfl_xor(m, 4));
      m = fmaxf(m, __shfl_xor(m, 8));
      mx[v] = m;
    }

    // p = exp(s - max) as f16 into LDS rows (row = local batch, col = j).
    // exp(x) = exp2(x * log2e) -> one v_mul + raw v_exp_f32.
    #pragma unroll
    for (int t = 0; t < 4; ++t)
      #pragma unroll
      for (int v = 0; v < 8; ++v) {
        int bl = v + 8 * hw;
        int j  = t * 16 + l16;
        *(_Float16*)(lds_raw + bl * ROWB + j * 2) =
            (_Float16)fast_exp2((s[t][v] - mx[v]) * LOG2E);
      }
    asm volatile("s_wait_dscnt 0" ::: "memory");  // in-wave LDS store->load fence

    // A fragments (M=16 x K=32 f16): lane row = l16; elements 0..15 <-> k =
    // kb + hw*16 .. +15 (contiguous, matching B). Each 8-half
    // run is 16 contiguous bytes -> ds_load_b128.
    h8 a0lo = *(const h8*)(lds_raw + l16 * ROWB + ( 0 + hw * 16) * 2);
    h8 a0hi = *(const h8*)(lds_raw + l16 * ROWB + ( 8 + hw * 16) * 2);
    h8 a1lo = *(const h8*)(lds_raw + l16 * ROWB + (32 + hw * 16) * 2);
    h8 a1hi = *(const h8*)(lds_raw + l16 * ROWB + (40 + hw * 16) * 2);
    v16h A0, A1;
    #pragma unroll
    for (int e = 0; e < 8; ++e) {
      A0[e] = a0lo[e]; A0[e + 8] = a0hi[e];
      A1[e] = a1lo[e]; A1[e + 8] = a1hi[e];
    }

    // q = p @ E : 4 j-tiles, K=64 split into two 16x16x32 WMMAs each.
    #pragma unroll
    for (int t = 0; t < 4; ++t) {
      v8f acc = {};
      acc = __builtin_amdgcn_wmma_f32_16x16x32_f16(false, A0, false, Bf[t][0],
                                                   (short)0, acc, false, false);
      acc = __builtin_amdgcn_wmma_f32_16x16x32_f16(false, A1, false, Bf[t][1],
                                                   (short)0, acc, false, false);
      #pragma unroll
      for (int v = 0; v < 8; ++v) {
        // q >= exp(-0.1) ~ 0.9 by construction (stabilized p has a 1.0 entry),
        // so raw v_log_f32 is safe: ln(q) = log2(q) * ln2.
        float nxt = fmaf(fast_log2(acc[v]), LN2, mx[v]) + ef[t][v];
        s[t][v] = fmaf(mf[v], nxt - s[t][v], s[t][v]);  // branchless mask blend
      }
    }
  }

  // denominator_b = logsumexp_j(s[b,j] + end_transitions[j])
  #pragma unroll
  for (int t = 0; t < 4; ++t) {
    float e = end_t[t * 16 + l16];
    #pragma unroll
    for (int v = 0; v < 8; ++v) s[t][v] += e;
  }
  #pragma unroll
  for (int v = 0; v < 8; ++v) {
    float m = fmaxf(fmaxf(s[0][v], s[1][v]), fmaxf(s[2][v], s[3][v]));
    m = fmaxf(m, __shfl_xor(m, 1));
    m = fmaxf(m, __shfl_xor(m, 2));
    m = fmaxf(m, __shfl_xor(m, 4));
    m = fmaxf(m, __shfl_xor(m, 8));
    float sm = fast_exp2((s[0][v] - m) * LOG2E) + fast_exp2((s[1][v] - m) * LOG2E) +
               fast_exp2((s[2][v] - m) * LOG2E) + fast_exp2((s[3][v] - m) * LOG2E);
    sm += __shfl_xor(sm, 1);
    sm += __shfl_xor(sm, 2);
    sm += __shfl_xor(sm, 4);
    sm += __shfl_xor(sm, 8);
    if (l16 == 0) den[b0 + v + 8 * hw] = fmaf(fast_log2(sm), LN2, m);
  }
}

// ---------------------------------------------------------------------------
// Kernel 3: final reduction -> scalar mean log-likelihood.
// ---------------------------------------------------------------------------
__global__ __launch_bounds__(256)
void crf_finalize(const float* __restrict__ num, const float* __restrict__ den,
                  const int* __restrict__ mask, float* __restrict__ out) {
  __shared__ float red[256];
  float acc = 0.f;
  for (int b = threadIdx.x; b < CRF_B; b += 256) acc += num[b] - den[b];
  red[threadIdx.x] = acc;
  __syncthreads();
  for (int s = 128; s > 0; s >>= 1) {
    if (threadIdx.x < s) red[threadIdx.x] += red[threadIdx.x + s];
    __syncthreads();
  }
  float llh = red[0];
  __syncthreads();
  float ms = 0.f;
  for (int i = threadIdx.x; i < CRF_B * CRF_S; i += 256) ms += (float)mask[i];
  red[threadIdx.x] = ms;
  __syncthreads();
  for (int s = 128; s > 0; s >>= 1) {
    if (threadIdx.x < s) red[threadIdx.x] += red[threadIdx.x + s];
    __syncthreads();
  }
  if (threadIdx.x == 0) out[0] = llh / red[0];
}

// ---------------------------------------------------------------------------
extern "C" void kernel_launch(void* const* d_in, const int* in_sizes, int n_in,
                              void* d_out, int out_size, void* d_ws, size_t ws_size,
                              hipStream_t stream) {
  const float* em    = (const float*)d_in[0];  // (B,S,T) f32
  const int*   tags  = (const int*)d_in[1];    // (B,S) int
  const int*   mask  = (const int*)d_in[2];    // (B,S) int32
  const float* st    = (const float*)d_in[3];  // (T,)
  const float* et    = (const float*)d_in[4];  // (T,)
  const float* tr    = (const float*)d_in[5];  // (T,T)

  float* num = (float*)d_ws;        // [B]
  float* den = num + CRF_B;         // [B]
  float* out = (float*)d_out;       // [1]

  crf_numerator<<<dim3((CRF_B + 255) / 256), dim3(256), 0, stream>>>(
      em, tags, mask, st, et, tr, num);
  crf_forward_scan<<<dim3(CRF_B / 16), dim3(32), 0, stream>>>(
      em, mask, st, et, tr, den);
  crf_finalize<<<dim3(1), dim3(256), 0, stream>>>(num, den, mask, out);
}
